// KernelSpecifiedResnetBlock_53772990546371
// MI455X (gfx1250) — matrix operand order repaired and checked by reading, not compiled, not run on hardware
//
#include <hip/hip_runtime.h>
#include <hip/hip_bf16.h>

typedef __attribute__((ext_vector_type(16))) _Float16 v16h;
typedef __attribute__((ext_vector_type(8)))  _Float16 v8h;
typedef __attribute__((ext_vector_type(8)))  float    v8f;
typedef __attribute__((ext_vector_type(4)))  int      v4i;

#define BN   16
#define CN   64
#define HW   128
#define KTOT 576          // Cin * 9
#define HWHW (HW*HW)      // 16384
#define NSTEP (KTOT/32)   // 18 K-steps

// ---------------------------------------------------------------------------
// CDNA5 async global->LDS (builtin confirmed present; param1 = AS1 int4*)
// ---------------------------------------------------------------------------
#if __has_builtin(__builtin_amdgcn_global_load_async_to_lds_b128)
#define HAVE_ASYNC_LDS 1
typedef __attribute__((address_space(1))) v4i* as1_v4i_p;
typedef __attribute__((address_space(3))) v4i* as3_v4i_p;
#else
#define HAVE_ASYNC_LDS 0
#endif

__device__ __forceinline__ void copy16_g2l(const _Float16* g, _Float16* l) {
#if HAVE_ASYNC_LDS
    __builtin_amdgcn_global_load_async_to_lds_b128(
        (as1_v4i_p)g, (as3_v4i_p)l, 0, 0);
#else
    *(v8h*)l = *(const v8h*)g;
#endif
}

__device__ __forceinline__ void wait_async() {
#if HAVE_ASYNC_LDS
    asm volatile("s_wait_asynccnt 0x0" ::: "memory");
#endif
}

// ---------------------------------------------------------------------------
// Elementwise helpers
// ---------------------------------------------------------------------------
__global__ __launch_bounds__(256) void zero_f32(float* p, int n) {
    int i = blockIdx.x * 256 + threadIdx.x;
    if (i < n) p[i] = 0.f;
}

__global__ __launch_bounds__(256) void to_half(const float* __restrict__ in,
                                               _Float16* __restrict__ out, int n) {
    int i = blockIdx.x * 256 + threadIdx.x;
    if (i < n) out[i] = (_Float16)in[i];
}

// ---------------------------------------------------------------------------
// Implicit-GEMM 3x3 conv via WMMA. One workgroup = one (batch b, row h).
// Double-buffered LDS; async A staging; transposed B slab so all fragment
// loads are ds_load_b128. Computes y[b,:,h,:] f32 + per-(b,c) sum/sumsq.
// ---------------------------------------------------------------------------
__global__ __launch_bounds__(256) void conv_wmma(
    const _Float16* __restrict__ xin,   // [B, C, H, W]   f16
    const _Float16* __restrict__ apack, // [B, Cout, 576] f16 weights
    float* __restrict__ yout,           // [B, C, H, W]   f32
    float* __restrict__ stats)          // [B, C, 2]      f32 (sum, sumsq)
{
    // stride 40 halves = 80 B (multiple of 16 B -> aligned b128 runs)
    __shared__ __align__(16) _Float16 As[2][64][40];    // 64 x 32 A-slab
    __shared__ __align__(16) _Float16 BsT[2][128][40];  // transposed 32x128 im2col
    __shared__ float sstat[64][2];

    const int h    = blockIdx.x;        // 0..127 image row
    const int b    = blockIdx.y;        // 0..15  batch
    const int tid  = threadIdx.x;
    const int lane = tid & 31;
    const int wave = tid >> 5;          // 0..7 -> N subtile
    const int hlf  = lane >> 4;         // 0/1
    const int l15  = lane & 15;

    if (tid < 128) sstat[tid >> 1][tid & 1] = 0.f;

    v8f acc[4];
#pragma unroll
    for (int m = 0; m < 4; ++m)
#pragma unroll
        for (int r = 0; r < 8; ++r) acc[m][r] = 0.f;

    const _Float16* ap = apack + (size_t)b * CN * KTOT;
    const _Float16* xb = xin + (size_t)b * CN * HWHW;

    // --- stage A-slab (64x32) for K-step k0: 256 x 16B async chunks ----------
    auto stageA = [&](int k0, int buf) {
        const int co = tid >> 2;          // 0..63
        const int c4 = (tid & 3) * 8;     // half-offset 0,8,16,24
        copy16_g2l(ap + (size_t)co * KTOT + k0 + c4, &As[buf][co][c4]);
    };
    // --- stage transposed im2col B-slab (BsT[w][kl]) for K-step k0 -----------
    auto stageB = [&](int k0, int buf) {
        const int kl = tid >> 3;          // 0..31
        const int w0 = (tid & 7) * 16;
        const int k  = k0 + kl;
        const int ci = k / 9;
        const int t  = k - ci * 9;
        const int kh = t / 3;
        const int kw = t - kh * 3;
        const int hs = h + kh - 1;
        const bool hv = (unsigned)hs < (unsigned)HW;
        const _Float16* row = xb + ((size_t)ci * HW + hs) * HW;
        if (hv) __builtin_prefetch(row + w0, 0, 1);   // global_prefetch_b8
#pragma unroll
        for (int i = 0; i < 16; ++i) {
            const int ws = w0 + i + kw - 1;
            _Float16 v = (hv && (unsigned)ws < (unsigned)HW)
                             ? row[ws] : (_Float16)0.f;
            BsT[buf][w0 + i][kl] = v;
        }
    };

    stageA(0, 0);
    stageB(0, 0);
    wait_async();
    __syncthreads();

    for (int s = 0; s < NSTEP; ++s) {
        const int buf = s & 1;
        if (s + 1 < NSTEP) {              // overlap next stage with compute
            stageA((s + 1) * 32, buf ^ 1);
            stageB((s + 1) * 32, buf ^ 1);
        }

        // B fragment: two contiguous 8-half runs at K = hlf*8 and 16+hlf*8 in BsT
        const int n = wave * 16 + l15;
        v16h bf;
        {
            const v8h b0 = *(const v8h*)&BsT[buf][n][hlf * 8];
            const v8h b1 = *(const v8h*)&BsT[buf][n][16 + hlf * 8];
#pragma unroll
            for (int j = 0; j < 8; ++j) { bf[j] = b0[j]; bf[8 + j] = b1[j]; }
        }

#pragma unroll
        for (int m = 0; m < 4; ++m) {
            const int row = m * 16 + l15;
            // A fragment: two contiguous 8-half runs at K = hlf*8 and 16+hlf*8
            const v8h a0 = *(const v8h*)&As[buf][row][hlf * 8];
            const v8h a1 = *(const v8h*)&As[buf][row][16 + hlf * 8];
            v16h af;
#pragma unroll
            for (int j = 0; j < 8; ++j) { af[j] = a0[j]; af[8 + j] = a1[j]; }
            acc[m] = __builtin_amdgcn_wmma_f32_16x16x32_f16(
                false, af, false, bf, (short)0, acc[m], false, false);
        }

        wait_async();
        __syncthreads();
    }

    // --- epilogue: store fp32 conv output + LDS stat reduction ---------------
#pragma unroll
    for (int m = 0; m < 4; ++m) {
#pragma unroll
        for (int r = 0; r < 8; ++r) {
            const int co = m * 16 + r + hlf * 8;   // D layout: M = r + 8*half
            const int wc = wave * 16 + l15;        // D layout: N = lane&15
            const float v = acc[m][r];
            yout[(((size_t)b * CN + co) * HW + h) * HW + wc] = v;
            atomicAdd(&sstat[co][0], v);
            atomicAdd(&sstat[co][1], v * v);
        }
    }
    __syncthreads();
    if (tid < 128)
        atomicAdd(&stats[((size_t)b * CN + (tid >> 1)) * 2 + (tid & 1)],
                  sstat[tid >> 1][tid & 1]);
}

// ---------------------------------------------------------------------------
// InstanceNorm + LeakyReLU -> f16 (input of conv2). One block per (b,c).
// ---------------------------------------------------------------------------
__global__ __launch_bounds__(256) void norm_lrelu_half(
    const float* __restrict__ y, const float* __restrict__ stats,
    _Float16* __restrict__ out)
{
    const int bc = blockIdx.x;
    const float inv = 1.f / (float)HWHW;
    const float mean = stats[bc * 2] * inv;
    const float var  = stats[bc * 2 + 1] * inv - mean * mean;
    const float sc   = rsqrtf(var + 1e-5f);
    const float* yp = y + (size_t)bc * HWHW;
    _Float16*    op = out + (size_t)bc * HWHW;
    for (int i = threadIdx.x; i < HWHW; i += 256) {
        float v = (yp[i] - mean) * sc;
        v = v >= 0.f ? v : 0.2f * v;
        op[i] = (_Float16)v;
    }
}

// ---------------------------------------------------------------------------
// Final: out = leaky_relu(x + instance_norm(y2)).  One block per (b,c).
// ---------------------------------------------------------------------------
__global__ __launch_bounds__(256) void final_residual(
    const float* __restrict__ x, const float* __restrict__ y,
    const float* __restrict__ stats, float* __restrict__ out)
{
    const int bc = blockIdx.x;
    const float inv = 1.f / (float)HWHW;
    const float mean = stats[bc * 2] * inv;
    const float var  = stats[bc * 2 + 1] * inv - mean * mean;
    const float sc   = rsqrtf(var + 1e-5f);
    const float* xp = x + (size_t)bc * HWHW;
    const float* yp = y + (size_t)bc * HWHW;
    float*       op = out + (size_t)bc * HWHW;
    for (int i = threadIdx.x; i < HWHW; i += 256) {
        float v = xp[i] + (yp[i] - mean) * sc;
        op[i] = v >= 0.f ? v : 0.2f * v;
    }
}

// ---------------------------------------------------------------------------
extern "C" void kernel_launch(void* const* d_in, const int* in_sizes, int n_in,
                              void* d_out, int out_size, void* d_ws, size_t ws_size,
                              hipStream_t stream) {
    (void)in_sizes; (void)n_in; (void)out_size; (void)ws_size;
    const float* x  = (const float*)d_in[0];
    const float* k1 = (const float*)d_in[1];
    const float* k2 = (const float*)d_in[2];
    float* out = (float*)d_out;

    char* ws = (char*)d_ws;
    size_t off = 0;
    const size_t NX = (size_t)BN * CN * HWHW;          // 16.78M elems
    const size_t NK = (size_t)BN * CN * KTOT;          // 589824 elems

    _Float16* xh = (_Float16*)(ws + off); off += NX * 2;
    _Float16* hh = (_Float16*)(ws + off); off += NX * 2;
    float*    yb = (float*)   (ws + off); off += NX * 4;
    _Float16* A1 = (_Float16*)(ws + off); off += NK * 2;
    _Float16* A2 = (_Float16*)(ws + off); off += NK * 2;
    float*    st = (float*)   (ws + off); off += 2 * BN * CN * 2 * 4;
    float* st1 = st;
    float* st2 = st + BN * CN * 2;

    // 0. zero stats (both convs)
    zero_f32<<<(2 * BN * CN * 2 + 255) / 256, 256, 0, stream>>>(st, 2 * BN * CN * 2);
    // 1. demote inputs / pack weights to f16 ([B,Co,Ci,3,3] flattens to [B,Co,576])
    to_half<<<(int)((NX + 255) / 256), 256, 0, stream>>>(x,  xh, (int)NX);
    to_half<<<(int)((NK + 255) / 256), 256, 0, stream>>>(k1, A1, (int)NK);
    to_half<<<(int)((NK + 255) / 256), 256, 0, stream>>>(k2, A2, (int)NK);
    // 2. conv1 (WMMA implicit GEMM) + stats
    conv_wmma<<<dim3(HW, BN), 256, 0, stream>>>(xh, A1, yb, st1);
    // 3. instance-norm + leaky-relu -> f16 intermediate
    norm_lrelu_half<<<BN * CN, 256, 0, stream>>>(yb, st1, hh);
    // 4. conv2 + stats
    conv_wmma<<<dim3(HW, BN), 256, 0, stream>>>(hh, A2, yb, st2);
    // 5. final residual + leaky-relu
    final_residual<<<BN * CN, 256, 0, stream>>>(x, yb, st2, out);
}
